// HybridSigLSTM_48473000902900
// MI455X (gfx1250) — matrix operand order repaired and checked by reading, not compiled, not run on hardware
//
#include <hip/hip_runtime.h>
#include <hip/hip_bf16.h>

// ---------------------------------------------------------------------------
// HybridSigLSTM on gfx1250 (MI455X, wave32, WMMA bf16 16x16x32)
//
// Persistent kernel: one block = 16 batch rows for all 256 timesteps.
// Weights live in LDS as bf16, pre-packed into the exact WMMA B-fragment
// per-lane order by pack_kernel. h0/h1/x are staged in LDS in the WMMA
// 16-bit A-matrix layout. prev_delta handled as a rank-1 C-fragment update
// with pd preloaded into registers. Gate scratch z is stored transposed
// (z[n][m]) so each lane's C column writes as two ds_store_b128.
// ---------------------------------------------------------------------------

typedef __attribute__((ext_vector_type(16))) __bf16 v16bf;
typedef __attribute__((ext_vector_type(8)))  float  v8f;

#define B_SZ   4096
#define S_LEN  256
#define DIN    4
#define H_DIM  50
#define NG     200     // 4*H
#define NGT    13      // ceil(200/16) N-tiles per gate matmul
#define NGP    208     // padded gate width
#define XSTAT  34      // static input dims (features + signature)

// tile table offsets (each tile = 256 dwords = 32x16 bf16 B-fragment)
#define T_WIH0 0
#define T_WHH0 26
#define T_WIH1 52
#define T_WHH1 78
#define T_W1   104
#define NTILES 108

__device__ __forceinline__ unsigned short f2bf(float x) {
    unsigned u = __float_as_uint(x);
    unsigned r = (u + 0x7FFFu + ((u >> 16) & 1u)) >> 16;   // RNE
    return (unsigned short)r;
}

__device__ __forceinline__ float sigf(float x) {
    x = fminf(fmaxf(x, -30.f), 30.f);
    float e = __expf(-x);
    return __builtin_amdgcn_rcpf(1.f + e);
}
__device__ __forceinline__ float tanh_f(float x) {
    x = fminf(fmaxf(x, -15.f), 15.f);
    float e = __expf(2.f * x);
    return (e - 1.f) * __builtin_amdgcn_rcpf(e + 1.f);
}

// --- WMMA fragment loaders -------------------------------------------------
// A layout (16-bit, 16x32 tile): lane L -> M = L&15, khalf = L>>4.
__device__ __forceinline__ v16bf loadA(const unsigned short* base, int lane, int kt) {
    int m  = lane & 15;
    int kh = lane >> 4;
    const unsigned short* p = base + m * 64 + kt * 32 + kh * 16;
    union { v16bf v; uint4 q[2]; } u;
    u.q[0] = *(const uint4*)(p);
    u.q[1] = *(const uint4*)(p + 8);
    return u.v;
}

// B fragments are pre-packed per lane: tile + lane*8 dwords (32B contiguous).
__device__ __forceinline__ v16bf loadB(const unsigned* tile, int lane) {
    union { v16bf v; uint4 q[2]; } u;
    const uint4* p = (const uint4*)(tile + lane * 8);
    u.q[0] = p[0];
    u.q[1] = p[1];
    return u.v;
}

__device__ __forceinline__ v8f wmma_bf16(v16bf a, v16bf b, v8f c) {
    return __builtin_amdgcn_wmma_f32_16x16x32_bf16(
        false, a, false, b, (short)0, c, false, false);
}

// --- gate matmul: z^T = A1*W1^T + A2*W2^T + bias [+ pd*wcol] ---------------
// zout is TRANSPOSED: zout[n * 16 + m]; each lane writes two float4s.
template <bool WITH_PD>
__device__ __forceinline__ void gate_matmul(
    const unsigned short* __restrict__ A1, const unsigned short* __restrict__ A2,
    const unsigned* __restrict__ W1base, const unsigned* __restrict__ W2base,
    const float* __restrict__ bias, const float* __restrict__ wcol,
    const float* __restrict__ pd, float* __restrict__ zout, int lane, int wid)
{
    // A fragments are nt-invariant: load once.
    v16bf a0 = loadA(A1, lane, 0);
    v16bf a1 = loadA(A1, lane, 1);
    v16bf a2 = loadA(A2, lane, 0);
    v16bf a3 = loadA(A2, lane, 1);

    const int nl = lane & 15;
    const int mh = (lane >> 4) * 8;

    float pdv[8];
    if (WITH_PD) {
        float4 p0 = *(const float4*)(pd + mh);
        float4 p1 = *(const float4*)(pd + mh + 4);
        pdv[0] = p0.x; pdv[1] = p0.y; pdv[2] = p0.z; pdv[3] = p0.w;
        pdv[4] = p1.x; pdv[5] = p1.y; pdv[6] = p1.z; pdv[7] = p1.w;
    }

    for (int nt = wid; nt < NGT; nt += 8) {            // scalar (SGPR) loop
        v8f acc = {};
        acc = wmma_bf16(a0, loadB(W1base + nt * 256, lane), acc);
        acc = wmma_bf16(a1, loadB(W1base + (NGT + nt) * 256, lane), acc);
        acc = wmma_bf16(a2, loadB(W2base + nt * 256, lane), acc);
        acc = wmma_bf16(a3, loadB(W2base + (NGT + nt) * 256, lane), acc);

        const int n  = nt * 16 + nl;
        const float bb = bias[n];
        const float wc = WITH_PD ? wcol[n] : 0.f;
        float r[8];
#pragma unroll
        for (int v = 0; v < 8; ++v) {
            r[v] = acc[v] + bb;
            if (WITH_PD) r[v] += pdv[v] * wc;
        }
        float4* zp = (float4*)(zout + n * 16 + mh);
        zp[0] = make_float4(r[0], r[1], r[2], r[3]);
        zp[1] = make_float4(r[4], r[5], r[6], r[7]);
    }
}

// ---------------------------------------------------------------------------
// pack_kernel: f32 weights -> bf16 WMMA B-fragment tiles + fused consts.
// B layout packed here, contiguous k to match loadA (bf16 32x16 tile): lane L -> N = L&15, khalf = L>>4,
// dword j holds K = khalf*16 + 2j (lo) and +1 (hi).
// ---------------------------------------------------------------------------
__global__ __launch_bounds__(256) void pack_kernel(
    const float* __restrict__ wih0, const float* __restrict__ whh0,
    const float* __restrict__ wih1, const float* __restrict__ whh1,
    const float* __restrict__ w1,
    const float* __restrict__ bih0, const float* __restrict__ bhh0,
    const float* __restrict__ bih1, const float* __restrict__ bhh1,
    const float* __restrict__ b1,   const float* __restrict__ w2,
    const float* __restrict__ b2,
    unsigned* __restrict__ wpack, float* __restrict__ consts)
{
    const int blk = blockIdx.x, tid = threadIdx.x;
    if (blk < NTILES) {
        const float* src; int stride, kused, nused, kt, nt;
        if (blk < 26)       { src = wih0; stride = 35; kused = 34; nused = 200; int l = blk;       kt = l / 13; nt = l % 13; }
        else if (blk < 52)  { src = whh0; stride = 50; kused = 50; nused = 200; int l = blk - 26;  kt = l / 13; nt = l % 13; }
        else if (blk < 78)  { src = wih1; stride = 50; kused = 50; nused = 200; int l = blk - 52;  kt = l / 13; nt = l % 13; }
        else if (blk < 104) { src = whh1; stride = 50; kused = 50; nused = 200; int l = blk - 78;  kt = l / 13; nt = l % 13; }
        else                { src = w1;   stride = 50; kused = 50; nused = 25;  int l = blk - 104; kt = l >> 1; nt = l & 1; }
        int lane = tid >> 3, j = tid & 7;
        int kh = lane >> 4, nl = lane & 15;
        int k0 = kt * 32 + kh * 16 + 2 * j;
        int n  = nt * 16 + nl;
        float a = 0.f, b = 0.f;
        if (n < nused) {
            if (k0     < kused) a = src[n * stride + k0];
            if (k0 + 1 < kused) b = src[n * stride + k0 + 1];
        }
        wpack[blk * 256 + lane * 8 + j] =
            (unsigned)f2bf(a) | ((unsigned)f2bf(b) << 16);
    } else {
        if (tid < 200) {
            consts[tid]       = bih0[tid] + bhh0[tid];
            consts[256 + tid] = bih1[tid] + bhh1[tid];
            consts[512 + tid] = wih0[tid * 35 + 34];   // prev_delta column
        }
        if (tid < 32) {
            consts[768 + tid] = (tid < 25) ? b1[tid] : 0.f;
            consts[800 + tid] = (tid < 25) ? w2[tid] : 0.f;
        }
        if (tid == 0) consts[832] = b2[0];
        if (tid >= 200 && tid < 256) { consts[tid] = 0.f; consts[256 + tid] = 0.f; consts[512 + tid] = 0.f; }
        if (tid > 832 && tid < 1024) consts[tid] = 0.f;
    }
}

// ---------------------------------------------------------------------------
// main persistent kernel: 256 blocks x 256 threads (8 waves), 16 rows/block
// ---------------------------------------------------------------------------
__global__ __launch_bounds__(256) void sig_lstm_kernel(
    const float* __restrict__ features, const unsigned* __restrict__ wpack,
    const float* __restrict__ consts, float* __restrict__ out)
{
    __shared__ unsigned s_w[NTILES * 256];                    // 108 KB bf16 weights
    __shared__ float    s_c[1024];                            // fused consts
    __shared__ __align__(16) float s_z[NGP * 16];             // gate scratch (transposed)
    __shared__ __align__(16) unsigned short s_x[16 * 64];     // x (A layout)
    __shared__ __align__(16) unsigned short s_h0[16 * 64];    // h0 (A layout)
    __shared__ __align__(16) unsigned short s_h1[16 * 64];    // h1 (A layout)
    __shared__ __align__(16) float s_pd[16];

    const int tid  = threadIdx.x;
    const int lane = tid & 31;
    // force scalar wave id -> all wave-uniform control flow becomes SALU
    const int wid  = __builtin_amdgcn_readfirstlane(tid >> 5);
    const int b0   = blockIdx.x * 16;

    // ---- one-time loads / init ----
    {
        const uint4* src = (const uint4*)wpack;
        uint4*       dst = (uint4*)s_w;
        for (int i = tid; i < NTILES * 64; i += 256) dst[i] = src[i];
        for (int i = tid; i < 1024; i += 256) s_c[i] = consts[i];
        unsigned* zx = (unsigned*)s_x; unsigned* z0 = (unsigned*)s_h0; unsigned* z1 = (unsigned*)s_h1;
        for (int i = tid; i < 512; i += 256) { zx[i] = 0u; z0[i] = 0u; z1[i] = 0u; }
        if (tid < 16) s_pd[tid] = 0.f;
    }
    __syncthreads();

    const float* b0s  = s_c;
    const float* b1s  = s_c + 256;
    const float* wcol = s_c + 512;
    const float* bh1  = s_c + 768;
    const float* w2v  = s_c + 800;
    const float  b2v  = s_c[832];

    float c0r[4] = {0.f, 0.f, 0.f, 0.f};
    float c1r[4] = {0.f, 0.f, 0.f, 0.f};
    float wreg[5][5];                                   // rolling aug window (rows 0..15)
    float4 fcur = {0.f, 0.f, 0.f, 0.f};
    if (tid < 16)
        fcur = *(const float4*)(features + (size_t)(b0 + tid) * S_LEN * DIN);

    for (int t = 0; t < S_LEN; ++t) {
        // ---- stage x_t = [features, lvl1 sig, lvl2 sig] as bf16 ----
        if (tid < 16) {
            float aug[5] = { t * (1.f / (S_LEN - 1)), fcur.x, fcur.y, fcur.z, fcur.w };
            if (t == 0) {
#pragma unroll
                for (int k = 0; k < 5; ++k)
#pragma unroll
                    for (int i = 0; i < 5; ++i) wreg[k][i] = aug[i];
            } else {
#pragma unroll
                for (int k = 0; k < 4; ++k)
#pragma unroll
                    for (int i = 0; i < 5; ++i) wreg[k][i] = wreg[k + 1][i];
#pragma unroll
                for (int i = 0; i < 5; ++i) wreg[4][i] = aug[i];
            }
            unsigned short* xr = s_x + tid * 64;
            xr[0] = f2bf(fcur.x); xr[1] = f2bf(fcur.y);
            xr[2] = f2bf(fcur.z); xr[3] = f2bf(fcur.w);
#pragma unroll
            for (int i = 0; i < 5; ++i) xr[4 + i] = f2bf(wreg[4][i] - wreg[0][i]);
#pragma unroll
            for (int i = 0; i < 5; ++i)
#pragma unroll
                for (int j = 0; j < 5; ++j) {
                    float s = 0.f;
#pragma unroll
                    for (int k = 0; k < 4; ++k)
                        s += (wreg[k + 1][i] - wreg[k][i]) * wreg[k][j];
                    xr[9 + i * 5 + j] = f2bf(s);
                }
            if (t + 1 < S_LEN)                           // prefetch next step
                fcur = *(const float4*)(features +
                        ((size_t)(b0 + tid) * S_LEN + (t + 1)) * DIN);
        }
        __syncthreads();

        // ---- LSTM layer 0 gates (with pd rank-1 update) ----
        gate_matmul<true>(s_x, s_h0, s_w + T_WIH0 * 256, s_w + T_WHH0 * 256,
                          b0s, wcol, s_pd, s_z, lane, wid);
        __syncthreads();

        // ---- layer 0 nonlinearity (z is transposed: z[g*16+m]) ----
#pragma unroll
        for (int r = 0; r < 4; ++r) {
            int idx = tid + r * 256;
            if (idx < 16 * H_DIM) {
                int m = idx / H_DIM, h = idx - m * H_DIM;
                float gi = s_z[h * 16 + m];
                float gf = s_z[(50 + h) * 16 + m];
                float gg = s_z[(100 + h) * 16 + m];
                float go = s_z[(150 + h) * 16 + m];
                float c  = sigf(gf) * c0r[r] + sigf(gi) * tanh_f(gg);
                c0r[r] = c;
                s_h0[m * 64 + h] = f2bf(sigf(go) * tanh_f(c));
            }
        }
        __syncthreads();

        // ---- LSTM layer 1 gates ----
        gate_matmul<false>(s_h0, s_h1, s_w + T_WIH1 * 256, s_w + T_WHH1 * 256,
                           b1s, nullptr, nullptr, s_z, lane, wid);
        __syncthreads();

        // ---- layer 1 nonlinearity ----
#pragma unroll
        for (int r = 0; r < 4; ++r) {
            int idx = tid + r * 256;
            if (idx < 16 * H_DIM) {
                int m = idx / H_DIM, h = idx - m * H_DIM;
                float gi = s_z[h * 16 + m];
                float gf = s_z[(50 + h) * 16 + m];
                float gg = s_z[(100 + h) * 16 + m];
                float go = s_z[(150 + h) * 16 + m];
                float c  = sigf(gf) * c1r[r] + sigf(gi) * tanh_f(gg);
                c1r[r] = c;
                s_h1[m * 64 + h] = f2bf(sigf(go) * tanh_f(c));
            }
        }
        __syncthreads();

        // ---- head: relu(h1 @ w1^T + b1), z transposed ----
        if (wid < 2) {                                   // scalar branch
            v8f acc = {};
            acc = wmma_bf16(loadA(s_h1, lane, 0),
                            loadB(s_w + (T_W1 + wid) * 256, lane), acc);
            acc = wmma_bf16(loadA(s_h1, lane, 1),
                            loadB(s_w + (T_W1 + 2 + wid) * 256, lane), acc);
            int n  = wid * 16 + (lane & 15);
            int mh = (lane >> 4) * 8;
            float bb = bh1[n];
            float r[8];
#pragma unroll
            for (int v = 0; v < 8; ++v) {
                float vv = acc[v] + bb;
                r[v] = vv > 0.f ? vv : 0.f;
            }
            float4* zp = (float4*)(s_z + n * 16 + mh);
            zp[0] = make_float4(r[0], r[1], r[2], r[3]);
            zp[1] = make_float4(r[4], r[5], r[6], r[7]);
        }
        __syncthreads();

        // ---- final projection, delta update, output ----
        if (tid < 16) {
            float raw = b2v;
#pragma unroll
            for (int j = 0; j < 25; ++j) raw += s_z[j * 16 + tid] * w2v[j];
            float inc = 0.2f * tanh_f(raw);
            float nd  = s_pd[tid] + inc;
            nd = fminf(fmaxf(nd, -1.5f), 1.5f);
            s_pd[tid] = nd;
            out[(size_t)(b0 + tid) * S_LEN + t] = nd;
        }
        __syncthreads();
    }
}

// ---------------------------------------------------------------------------
extern "C" void kernel_launch(void* const* d_in, const int* in_sizes, int n_in,
                              void* d_out, int out_size, void* d_ws, size_t ws_size,
                              hipStream_t stream)
{
    const float* features = (const float*)d_in[0];
    const float* wih0 = (const float*)d_in[1];
    const float* whh0 = (const float*)d_in[2];
    const float* bih0 = (const float*)d_in[3];
    const float* bhh0 = (const float*)d_in[4];
    const float* wih1 = (const float*)d_in[5];
    const float* whh1 = (const float*)d_in[6];
    const float* bih1 = (const float*)d_in[7];
    const float* bhh1 = (const float*)d_in[8];
    const float* w1   = (const float*)d_in[9];
    const float* b1   = (const float*)d_in[10];
    const float* w2   = (const float*)d_in[11];
    const float* b2   = (const float*)d_in[12];

    unsigned* wpack = (unsigned*)d_ws;
    float*    cst   = (float*)((char*)d_ws + (size_t)NTILES * 1024);

    pack_kernel<<<NTILES + 1, 256, 0, stream>>>(
        wih0, whh0, wih1, whh1, w1, bih0, bhh0, bih1, bhh1, b1, w2, b2,
        wpack, cst);

    sig_lstm_kernel<<<B_SZ / 16, 256, 0, stream>>>(
        features, wpack, cst, (float*)d_out);
}
